// L2OOptimizer_59657095741730
// MI455X (gfx1250) — matrix operand order repaired and checked by reading, not compiled, not run on hardware
//
#include <hip/hip_runtime.h>
#include <hip/hip_bf16.h>

#define HID 256
#define KTOT 288      // 256 hidden + 10 feat + 22 zero pad
#define SAMP 64       // samples per block
#define LSTRIDE 296   // LDS row stride in halves (288 + 8 pad)

typedef _Float16 half4v __attribute__((ext_vector_type(4)));
typedef _Float16 half8  __attribute__((ext_vector_type(8)));
typedef _Float16 v16h   __attribute__((ext_vector_type(16)));
typedef float    v8f    __attribute__((ext_vector_type(8)));
typedef float    f4     __attribute__((ext_vector_type(4)));

__device__ __forceinline__ v16h cat16(half8 lo, half8 hi) {
    return __builtin_shufflevector(lo, hi, 0,1,2,3,4,5,6,7,8,9,10,11,12,13,14,15);
}

// CDNA5 hardware tanh (TRANS32 op). The ISA requires >=1 instruction between a
// TRANS op and the first use of its result; the compiler can't see that an
// inline-asm op is TRANS, so we embed the documented s_delay_alu spacer.
__device__ __forceinline__ float fast_tanh(float x) {
    float r;
    asm("v_tanh_f32 %0, %1\n\t"
        "s_delay_alu instid0(TRANS32_DEP_1)"
        : "=v"(r) : "v"(x));
    return r;
}

// Pack A = [W_hh | W_ih | 0] (256 x 288) into f16; bias = b_ih + b_hh.
__global__ void l2o_prep(const float* __restrict__ W_hh, const float* __restrict__ W_ih,
                         const float* __restrict__ b_ih, const float* __restrict__ b_hh,
                         _Float16* __restrict__ Af16, float* __restrict__ bias) {
    const int row = blockIdx.x;      // 256 blocks
    const int col = threadIdx.x;     // 288 threads
    float v;
    if (col < 256)      v = W_hh[row * 256 + col];
    else if (col < 266) v = W_ih[row * 10 + (col - 256)];
    else                v = 0.0f;
    Af16[row * KTOT + col] = (_Float16)v;
    if (col == 0) bias[row] = b_ih[row] + b_hh[row];
}

__launch_bounds__(256)
__global__ void l2o_main(const float* __restrict__ x, const float* __restrict__ grad,
                         const float* __restrict__ hidden,
                         const float* __restrict__ W1, const float* __restrict__ b1,
                         const float* __restrict__ W2, const float* __restrict__ b2,
                         const float* __restrict__ fc_w, const float* __restrict__ fc_b,
                         const _Float16* __restrict__ Af16, const float* __restrict__ bias,
                         float* __restrict__ x_new, float* __restrict__ h_new,
                         float* __restrict__ delta_out) {
    __shared__ _Float16 Bt[SAMP * LSTRIDE];
    __shared__ float ldsDelta[SAMP];

    const int tid  = threadIdx.x;
    const int lane = tid & 31;
    const int wave = tid >> 5;
    const long sbase = (long)blockIdx.x * SAMP;

    // ---- stage hidden fp32 -> f16 into LDS (B operand, row = sample) ----
    for (int i = tid; i < SAMP * (HID / 4); i += 256) {
        const int s = i >> 6;       // sample (64 float4 chunks per sample)
        const int c = i & 63;
        const f4 v = *(const f4*)(hidden + (sbase + s) * HID + c * 4);
        half4v h;
        h[0] = (_Float16)v[0]; h[1] = (_Float16)v[1];
        h[2] = (_Float16)v[2]; h[3] = (_Float16)v[3];
        *(half4v*)(&Bt[s * LSTRIDE + c * 4]) = h;
    }

    // ---- tiny MLP front-end: feat -> LDS cols 256..287 ----
    if (tid < SAMP) {
        const long sample = sbase + tid;
        const float g = grad[sample];
        const float sg = (g > 0.0f) ? 1.0f : ((g < 0.0f) ? -1.0f : 0.0f);
        const float mg = logf(fabsf(g) + 1e-14f);
        float h1[10];
        #pragma unroll
        for (int j = 0; j < 10; ++j)
            h1[j] = fast_tanh(W1[2 * j] * sg + W1[2 * j + 1] * mg + b1[j]);
        _Float16* fp = &Bt[tid * LSTRIDE + 256];
        #pragma unroll
        for (int o = 0; o < 10; ++o) {
            float a = b2[o];
            #pragma unroll
            for (int j = 0; j < 10; ++j) a += W2[o * 10 + j] * h1[j];
            fp[o] = (_Float16)a;
        }
        #pragma unroll
        for (int o = 10; o < 32; ++o) fp[o] = (_Float16)0.0f;
        ldsDelta[tid] = 0.0f;
    }
    __syncthreads();

    // ---- GEMM: D[256 x 64] = A[256 x 288] x B[288 x 64] ----
    const int rowBase = wave * 32;                 // 2 row tiles per wave
    const int arow    = rowBase + (lane & 15);
    const int aoff    = (lane < 16) ? 0 : 8;       // 16-bit A layout split
    const int boff    = (lane < 16) ? 0 : 8;       // 16-bit B layout split
    const int scol    = lane & 15;

    v8f acc[2][4];
    #pragma unroll
    for (int rt = 0; rt < 2; ++rt)
        #pragma unroll
        for (int st = 0; st < 4; ++st)
            acc[rt][st] = (v8f){0.f,0.f,0.f,0.f,0.f,0.f,0.f,0.f};

    for (int kk = 0; kk < KTOT; kk += 32) {
        const _Float16* ap0 = Af16 + (long)arow * KTOT + kk + aoff;
        const _Float16* ap1 = Af16 + (long)(arow + 16) * KTOT + kk + aoff;
        const v16h a0 = cat16(*(const half8*)ap0, *(const half8*)(ap0 + 16));
        const v16h a1 = cat16(*(const half8*)ap1, *(const half8*)(ap1 + 16));
        #pragma unroll
        for (int st = 0; st < 4; ++st) {
            const _Float16* bp = &Bt[(st * 16 + scol) * LSTRIDE + kk + boff];
            const v16h b = cat16(*(const half8*)bp, *(const half8*)(bp + 16));
            acc[0][st] = __builtin_amdgcn_wmma_f32_16x16x32_f16(
                false, a0, false, b, (short)0, acc[0][st], false, false);
            acc[1][st] = __builtin_amdgcn_wmma_f32_16x16x32_f16(
                false, a1, false, b, (short)0, acc[1][st], false, false);
        }
    }

    // ---- epilogue: bias + tanh, store h_new, partial fc dot ----
    float dp[4] = {0.f, 0.f, 0.f, 0.f};
    #pragma unroll
    for (int rt = 0; rt < 2; ++rt) {
        const int rowB = rowBase + rt * 16 + ((lane < 16) ? 0 : 8);
        const f4 bA = *(const f4*)(bias + rowB);
        const f4 bB = *(const f4*)(bias + rowB + 4);
        const f4 wA = *(const f4*)(fc_w + rowB);
        const f4 wB = *(const f4*)(fc_w + rowB + 4);
        #pragma unroll
        for (int st = 0; st < 4; ++st) {
            f4 o0, o1;
            #pragma unroll
            for (int r = 0; r < 4; ++r) o0[r] = fast_tanh(acc[rt][st][r] + bA[r]);
            #pragma unroll
            for (int r = 0; r < 4; ++r) o1[r] = fast_tanh(acc[rt][st][4 + r] + bB[r]);
            float d = 0.0f;
            #pragma unroll
            for (int r = 0; r < 4; ++r) d += wA[r] * o0[r] + wB[r] * o1[r];
            dp[st] += d;
            float* hp = h_new + (sbase + st * 16 + scol) * (long)HID + rowB;
            *(f4*)hp = o0;
            *((f4*)hp + 1) = o1;
        }
    }
    #pragma unroll
    for (int st = 0; st < 4; ++st)
        atomicAdd(&ldsDelta[st * 16 + scol], dp[st]);
    __syncthreads();

    if (tid < SAMP) {
        const long sample = sbase + tid;
        const float d = ldsDelta[tid] + fc_b[0];
        delta_out[sample] = d;
        x_new[sample]    = x[sample] + d;
    }
}

extern "C" void kernel_launch(void* const* d_in, const int* in_sizes, int n_in,
                              void* d_out, int out_size, void* d_ws, size_t ws_size,
                              hipStream_t stream) {
    const float* x      = (const float*)d_in[0];
    const float* grad   = (const float*)d_in[1];
    const float* hidden = (const float*)d_in[2];
    const float* W1     = (const float*)d_in[3];
    const float* b1     = (const float*)d_in[4];
    const float* W2     = (const float*)d_in[5];
    const float* b2     = (const float*)d_in[6];
    const float* W_ih   = (const float*)d_in[7];
    const float* b_ih   = (const float*)d_in[8];
    const float* W_hh   = (const float*)d_in[9];
    const float* b_hh   = (const float*)d_in[10];
    const float* fc_w   = (const float*)d_in[11];
    const float* fc_b   = (const float*)d_in[12];

    const int M = in_sizes[0];   // 262144 samples

    _Float16* Af16 = (_Float16*)d_ws;
    float*    bias = (float*)((char*)d_ws + (size_t)HID * KTOT * sizeof(_Float16));

    float* x_new = (float*)d_out;
    float* h_new = x_new + M;
    float* delta = h_new + (long)M * HID;

    l2o_prep<<<HID, KTOT, 0, stream>>>(W_hh, W_ih, b_ih, b_hh, Af16, bias);
    l2o_main<<<M / SAMP, 256, 0, stream>>>(x, grad, hidden, W1, b1, W2, b2,
                                           fc_w, fc_b, Af16, bias,
                                           x_new, h_new, delta);
}
